// Rel_RNN_86904368267351
// MI455X (gfx1250) — matrix operand order repaired and checked by reading, not compiled, not run on hardware
//
#include <hip/hip_runtime.h>
#include <hip/hip_bf16.h>
#include <math.h>

typedef _Float16 half_t;
typedef __attribute__((ext_vector_type(16))) _Float16 v16h;
typedef __attribute__((ext_vector_type(8)))  _Float16 v8h;
typedef __attribute__((ext_vector_type(8)))  float    v8f;
typedef __attribute__((ext_vector_type(16))) float    v16f;

// D = A(16x32 f16) * B(32x16 f16) + C(16x16 f32)
__device__ __forceinline__ v8f wmma16x16x32(v16h a, v16h b, v8f c) {
  return __builtin_amdgcn_wmma_f32_16x16x32_f16(false, a, false, b, (short)0, c, false, false);
}

// A-matrix 16x32 f16 fragment from a padded row-major weight row (Kpad mult of 32).
// Layout: lanes 0-15: K = {0..15}; lanes 16-31: K = {16..31} (+kbase), contiguous.
__device__ __forceinline__ v16h load_a_frag(const half_t* __restrict__ wrow, int kbase, int lane) {
  const int off = (lane >= 16) ? 16 : 0;
  v8h lo = *(const v8h*)(wrow + kbase + off);
  v8h hi = *(const v8h*)(wrow + kbase + 8 + off);
  v16h a;
#pragma unroll
  for (int i = 0; i < 8; ++i) { a[i] = lo[i]; a[i + 8] = hi[i]; }
  return a;
}

// Build B fragment (32x16 f16) from a 64B-aligned f32 LDS row: K contiguous per lane.
__device__ __forceinline__ v16h load_b_frag_lds_f32(const float* __restrict__ row) {
  v16f hv = *(const v16f*)row;
  v16h b;
#pragma unroll
  for (int i = 0; i < 16; ++i) b[i] = (half_t)hv[i];
  return b;
}

// ---------------------------------------------------------------------------
// Kernel 0: convert weights f32 -> zero-padded f16 images for WMMA A operands.
// ---------------------------------------------------------------------------
#define S_C1 (64 * 64)          // conv1: 64 x (54 -> 64)
#define S_C2 (64 * 576)         // conv2: 64 x 576 (exact)
#define S_R  (112 * 416)        // R_w:  100->112 x 400->416
#define S_O  (128 * 416)        // O_w:  128 x 400->416
#define S_Q  (112 * 4224)       // Q_w:  100->112 x 4196->4224
#define S_TOTAL (S_C1 + S_C2 + S_R + S_O + S_Q)

__global__ void convert_weights_kernel(const float* __restrict__ c1w, const float* __restrict__ c2w,
                                       const float* __restrict__ Rw,  const float* __restrict__ Ow,
                                       const float* __restrict__ Qw,
                                       half_t* __restrict__ c1w16, half_t* __restrict__ c2w16,
                                       half_t* __restrict__ Rw16,  half_t* __restrict__ Ow16,
                                       half_t* __restrict__ Qw16) {
  for (int idx = blockIdx.x * blockDim.x + threadIdx.x; idx < S_TOTAL;
       idx += gridDim.x * blockDim.x) {
    int r = idx;
    if (r < S_C1) {
      int m = r / 64, k = r % 64;
      c1w16[r] = (half_t)((k < 54) ? c1w[m * 54 + k] : 0.f);
      continue;
    }
    r -= S_C1;
    if (r < S_C2) {
      c2w16[r] = (half_t)c2w[r];
      continue;
    }
    r -= S_C2;
    if (r < S_R) {
      int m = r / 416, k = r % 416;
      Rw16[r] = (half_t)((m < 100 && k < 400) ? Rw[m * 400 + k] : 0.f);
      continue;
    }
    r -= S_R;
    if (r < S_O) {
      int m = r / 416, k = r % 416;
      Ow16[r] = (half_t)((k < 400) ? Ow[m * 400 + k] : 0.f);
      continue;
    }
    r -= S_O;
    {
      int m = r / 4224, k = r % 4224;
      Qw16[r] = (half_t)((m < 100 && k < 4196) ? Qw[m * 4196 + k] : 0.f);
    }
  }
}

// Per-wave end-of-kernel BN stat flush: reduce 8 channel partials across each
// 16-lane half (channels are fixed per lane), then 2 atomics per channel row.
__device__ __forceinline__ void flush_stats(const float* ssum, const float* ssq,
                                            int mtile, int khalf, int nloc,
                                            float* __restrict__ stats) {
#pragma unroll
  for (int vv = 0; vv < 8; ++vv) {
    const int m = mtile * 16 + vv + (khalf ? 8 : 0);
    float s = ssum[vv], s2 = ssq[vv];
#pragma unroll
    for (int o = 1; o < 16; o <<= 1) {
      s  += __shfl_xor(s,  o, 16);
      s2 += __shfl_xor(s2, o, 16);
    }
    if (nloc == 0) {
      atomicAdd(&stats[m], s);
      atomicAdd(&stats[64 + m], s2);
    }
  }
}

// ---------------------------------------------------------------------------
// Kernel 1: conv1 (6->64ch, 3x3, pad 1) implicit GEMM with WMMA.
// One block per image. Input staged in LDS with zero halo; 2 dummy channels so
// the padded K range [54,64) reads zeros -> fully branchless B fragments.
// BN sum/sumsq accumulate in registers; one shuffle-reduce + atomic per wave.
// ---------------------------------------------------------------------------
#define T1_CH 8
#define T1_W  34
#define T1_SZ (T1_CH * T1_W * T1_W)   // 9248 halfs

__global__ void __launch_bounds__(256)
conv1_kernel(const float* __restrict__ x1, const float* __restrict__ x2,
             const half_t* __restrict__ w16, const float* __restrict__ bias,
             half_t* __restrict__ preBN, float* __restrict__ stats /* [64]sum,[64]sumsq */) {
  __shared__ half_t tile[T1_SZ];
  const int img = blockIdx.x;            // 0..1023  (b*64+t)
  const int tid = threadIdx.x;

  for (int i = tid; i < T1_SZ; i += 256) tile[i] = (half_t)0.f;
  __syncthreads();
  for (int i = tid; i < 6 * 1024; i += 256) {
    const int c = i >> 10, rem = i & 1023;
    const int h = rem >> 5, w = rem & 31;
    const float* src = (c < 3) ? x1 : x2;
    const int cc = (c < 3) ? c : c - 3;
    tile[c * (T1_W * T1_W) + (h + 1) * T1_W + (w + 1)] =
        (half_t)src[((size_t)img * 3 + cc) * 1024 + rem];
  }
  __syncthreads();

  const int lane  = tid & 31;
  const int wave  = tid >> 5;
  const int nloc  = lane & 15;
  const int khalf = lane >> 4;
  const int mtile = wave & 3;            // fixed channel tile per wave
  const int shalf = wave >> 2;

  // Hoisted A fragments (K = 64 -> 2 k-steps) and bias values (channels fixed)
  const half_t* arow = w16 + (mtile * 16 + nloc) * 64;
  const v16h a0 = load_a_frag(arow, 0, lane);
  const v16h a1 = load_a_frag(arow, 32, lane);
  float bvals[8], ssum[8] = {}, ssq[8] = {};
#pragma unroll
  for (int vv = 0; vv < 8; ++vv) bvals[vv] = bias[mtile * 16 + vv + (khalf ? 8 : 0)];

  for (int stile = shalf * 32; stile < shalf * 32 + 32; ++stile) {
    const int p  = stile * 16 + nloc;    // position 0..1023
    const int py = p >> 5, px = p & 31;
    const half_t* tbase = tile + py * T1_W + px;

    v8f acc = {};
#pragma unroll
    for (int ks = 0; ks < 2; ++ks) {
      v16h b;
#pragma unroll
      for (int e = 0; e < 16; ++e) {
        const int k  = ks * 32 + khalf * 16 + e;   // 0..63 (54..63 read zero ch)
        const int ci = k / 9, rr = k - ci * 9;
        b[e] = tbase[ci * (T1_W * T1_W) + (rr / 3) * T1_W + (rr % 3)];
      }
      acc = wmma16x16x32((ks == 0) ? a0 : a1, b, acc);
    }
    half_t* obase = preBN + ((size_t)img * 64 + mtile * 16 + (khalf ? 8 : 0)) * 1024 + p;
#pragma unroll
    for (int vv = 0; vv < 8; ++vv) {
      const float val = acc[vv] + bvals[vv];
      obase[(size_t)vv * 1024] = (half_t)val;
      ssum[vv] += val;
      ssq[vv]  += val * val;
    }
  }
  flush_stats(ssum, ssq, mtile, khalf, nloc, stats);
}

// ---------------------------------------------------------------------------
// Kernel 2: conv2 (64->64ch, 3x3, pad 1) implicit GEMM with WMMA (K=576 exact).
// Input staged in LDS with zero halo -> branchless B fragments.
// ---------------------------------------------------------------------------
#define T2_W  18
#define T2_SZ (64 * T2_W * T2_W)       // 20736 halfs

__global__ void __launch_bounds__(256)
conv2_kernel(const half_t* __restrict__ xin /* (1024,64,16,16) */,
             const half_t* __restrict__ w16, const float* __restrict__ bias,
             half_t* __restrict__ preBN, float* __restrict__ stats) {
  __shared__ half_t tile[T2_SZ];
  const int img = blockIdx.x;
  const int tid = threadIdx.x;

  for (int i = tid; i < T2_SZ; i += 256) tile[i] = (half_t)0.f;
  __syncthreads();
  for (int i = tid; i < 64 * 256; i += 256) {
    const int c = i >> 8, rem = i & 255;
    const int h = rem >> 4, w = rem & 15;
    tile[c * (T2_W * T2_W) + (h + 1) * T2_W + (w + 1)] =
        xin[((size_t)img * 64 + c) * 256 + rem];
  }
  __syncthreads();

  const int lane  = tid & 31;
  const int wave  = tid >> 5;
  const int nloc  = lane & 15;
  const int khalf = lane >> 4;
  const int mtile = wave & 3;
  const int shalf = wave >> 2;

  const half_t* arow = w16 + (mtile * 16 + nloc) * 576;
  float bvals[8], ssum[8] = {}, ssq[8] = {};
#pragma unroll
  for (int vv = 0; vv < 8; ++vv) bvals[vv] = bias[mtile * 16 + vv + (khalf ? 8 : 0)];

  for (int stile = shalf * 8; stile < shalf * 8 + 8; ++stile) {
    const int p  = stile * 16 + nloc;    // 0..255
    const int py = p >> 4, px = p & 15;
    const half_t* tbase = tile + py * T2_W + px;

    v8f acc = {};
    for (int ks = 0; ks < 18; ++ks) {
      v16h a = load_a_frag(arow, ks * 32, lane);
      v16h b;
#pragma unroll
      for (int e = 0; e < 16; ++e) {
        const int k  = ks * 32 + khalf * 16 + e;   // < 576 always
        const int ci = k / 9, rr = k - ci * 9;
        b[e] = tbase[ci * (T2_W * T2_W) + (rr / 3) * T2_W + (rr % 3)];
      }
      acc = wmma16x16x32(a, b, acc);
    }
    half_t* obase = preBN + ((size_t)img * 64 + mtile * 16 + (khalf ? 8 : 0)) * 256 + p;
#pragma unroll
    for (int vv = 0; vv < 8; ++vv) {
      const float val = acc[vv] + bvals[vv];
      obase[(size_t)vv * 256] = (half_t)val;
      ssum[vv] += val;
      ssq[vv]  += val * val;
    }
  }
  flush_stats(ssum, ssq, mtile, khalf, nloc, stats);
}

// ---------------------------------------------------------------------------
// Kernel 3: BatchNorm (batch stats) + ReLU + MaxPool2d(2). Generic.
// Output index ordering (img, c, ph, pw) matches both pooled1 and feats.
// ---------------------------------------------------------------------------
__global__ void bn_relu_pool_kernel(const half_t* __restrict__ preBN, half_t* __restrict__ outp,
                                    const float* __restrict__ stats,
                                    const float* __restrict__ g, const float* __restrict__ beta,
                                    float invN, int HW_in, int HW_out, int total) {
  for (int idx = blockIdx.x * blockDim.x + threadIdx.x; idx < total;
       idx += gridDim.x * blockDim.x) {
    const int pw  = idx % HW_out;
    const int ph  = (idx / HW_out) % HW_out;
    const int c   = (idx / (HW_out * HW_out)) % 64;
    const int img = idx / (HW_out * HW_out * 64);
    const float mean  = stats[c] * invN;
    const float var   = stats[64 + c] * invN - mean * mean;
    const float scale = g[c] * rsqrtf(var + 1e-5f);
    const float shift = beta[c] - mean * scale;
    const half_t* base = preBN + ((size_t)img * 64 + c) * HW_in * HW_in;
    float mx = -1e30f;
#pragma unroll
    for (int dy = 0; dy < 2; ++dy)
#pragma unroll
      for (int dx = 0; dx < 2; ++dx) {
        const float x = (float)base[(2 * ph + dy) * HW_in + (2 * pw + dx)];
        mx = fmaxf(mx, fmaxf(x * scale + shift, 0.f));
      }
    outp[idx] = (half_t)mx;
  }
}

// ---------------------------------------------------------------------------
// Kernel 4: sequential leaky-integrator RNN over T=64 + head. Single WG,
// 7 waves (one per 16-row M-tile), batch N=16 = one WMMA tile. Ha/Rt rows are
// padded (416 / 128) with zero tails so all B fragments are unguarded 64-byte
// vector LDS reads. Q-weight stream is prefetched (global_prefetch_b8).
// ---------------------------------------------------------------------------
__global__ void __launch_bounds__(224)
rnn_head_kernel(const half_t* __restrict__ feats /* (16*64, 4096) */,
                const half_t* __restrict__ Rw16, const half_t* __restrict__ Qw16,
                const half_t* __restrict__ Ow16,
                const float* __restrict__ Rb, const float* __restrict__ Qb,
                const float* __restrict__ Ob,
                const float* __restrict__ fc1w, const float* __restrict__ fc1b,
                const float* __restrict__ fc2w, const float* __restrict__ fc2b,
                float* __restrict__ out) {
  __shared__ __align__(64) float Ha[16 * 416];   // (b, a*100+q), zero-padded tail
  __shared__ __align__(64) float Rt[16 * 128];   // zero-padded tail
  __shared__ float OutL[16 * 128];
  __shared__ float Hh[16 * 64];

  const int tid   = threadIdx.x;
  const int lane  = tid & 31;
  const int wave  = tid >> 5;       // 0..6 -> M-tile
  const int nloc  = lane & 15;      // batch index (N)
  const int khalf = lane >> 4;
  const float Ac[4] = {0.f, 0.25f, 0.5f, 0.95f};

  for (int i = tid; i < 16 * 416; i += blockDim.x) Ha[i] = 0.f;
  for (int i = tid; i < 16 * 128; i += blockDim.x) Rt[i] = 0.f;
  __syncthreads();

  for (int t = 0; t < 64; ++t) {
    // Rt = tanh(Hf @ R_w^T + R_b):  M=100(pad112), K=400(pad416), N=16
    {
      v8f acc = {};
      const half_t* arow = Rw16 + (wave * 16 + nloc) * 416;
      const float*  hrow = &Ha[nloc * 416 + khalf * 16];
      for (int ks = 0; ks < 13; ++ks) {
        v16h a = load_a_frag(arow, ks * 32, lane);
        v16h b = load_b_frag_lds_f32(hrow + ks * 32);
        acc = wmma16x16x32(a, b, acc);
      }
#pragma unroll
      for (int vv = 0; vv < 8; ++vv) {
        const int m = wave * 16 + vv + (khalf ? 8 : 0);
        if (m < 100) Rt[nloc * 128 + m] = tanhf(acc[vv] + Rb[m]);
      }
    }
    __syncthreads();  // Rt ready; all waves done reading Ha
    // Qt = tanh([feat_t, Rt] @ Q_w^T + Q_b): K = 4096 + 100 (pad 4224)
    {
      v8f acc = {};
      const half_t* arow = Qw16 + (size_t)(wave * 16 + nloc) * 4224;
      const half_t* frow = feats + ((size_t)nloc * 64 + t) * 4096 + khalf * 16;
      for (int ks = 0; ks < 128; ++ks) {
        __builtin_prefetch(arow + (ks + 8) * 32, 0, 1);   // stream Q weights ahead
        v16h a = load_a_frag(arow, ks * 32, lane);
        v16h b = *(const v16h*)(frow + ks * 32);          // contiguous K, 32B aligned
        acc = wmma16x16x32(a, b, acc);
      }
      const float* rrow = &Rt[nloc * 128 + khalf * 16];
#pragma unroll
      for (int ks = 128; ks < 132; ++ks) {
        v16h a = load_a_frag(arow, ks * 32, lane);
        v16h b = load_b_frag_lds_f32(rrow + (ks - 128) * 32);
        acc = wmma16x16x32(a, b, acc);
      }
      // Ha = A*Ha + (1-A)*Qt  (each (b,m) owned by exactly one lane)
#pragma unroll
      for (int vv = 0; vv < 8; ++vv) {
        const int m = wave * 16 + vv + (khalf ? 8 : 0);
        if (m < 100) {
          const float qt = tanhf(acc[vv] + Qb[m]);
#pragma unroll
          for (int a2 = 0; a2 < 4; ++a2) {
            const int hidx = nloc * 416 + a2 * 100 + m;
            Ha[hidx] = Ac[a2] * Ha[hidx] + (1.f - Ac[a2]) * qt;
          }
        }
      }
    }
    __syncthreads();  // Ha updated before next step's reads
  }

  // out_last = tanh(Ha_flat @ O_w^T + O_b): M=128 -> 8 tiles over 7 waves
  for (int mt = wave; mt < 8; mt += 7) {
    v8f acc = {};
    const half_t* arow = Ow16 + (mt * 16 + nloc) * 416;
    const float*  hrow = &Ha[nloc * 416 + khalf * 16];
    for (int ks = 0; ks < 13; ++ks) {
      v16h a = load_a_frag(arow, ks * 32, lane);
      v16h b = load_b_frag_lds_f32(hrow + ks * 32);
      acc = wmma16x16x32(a, b, acc);
    }
#pragma unroll
    for (int vv = 0; vv < 8; ++vv) {
      const int m = mt * 16 + vv + (khalf ? 8 : 0);
      OutL[nloc * 128 + m] = tanhf(acc[vv] + Ob[m]);
    }
  }
  __syncthreads();
  // fc1 (relu) then fc2 (sigmoid) — tiny, scalar
  for (int i = tid; i < 16 * 64; i += blockDim.x) {
    const int b = i >> 6, j = i & 63;
    float s = fc1b[j];
    for (int q = 0; q < 128; ++q) s += OutL[b * 128 + q] * fc1w[j * 128 + q];
    Hh[i] = fmaxf(s, 0.f);
  }
  __syncthreads();
  for (int i = tid; i < 16; i += blockDim.x) {
    float s = fc2b[0];
    for (int q = 0; q < 64; ++q) s += Hh[i * 64 + q] * fc2w[q];
    out[i] = 1.f / (1.f + expf(-s));
  }
}

// ---------------------------------------------------------------------------
extern "C" void kernel_launch(void* const* d_in, const int* in_sizes, int n_in,
                              void* d_out, int out_size, void* d_ws, size_t ws_size,
                              hipStream_t stream) {
  const float* x1    = (const float*)d_in[0];
  const float* x2    = (const float*)d_in[1];
  const float* c1w   = (const float*)d_in[2];
  const float* c1b   = (const float*)d_in[3];
  const float* bn1g  = (const float*)d_in[4];
  const float* bn1b  = (const float*)d_in[5];
  const float* c2w   = (const float*)d_in[6];
  const float* c2b   = (const float*)d_in[7];
  const float* bn2g  = (const float*)d_in[8];
  const float* bn2b  = (const float*)d_in[9];
  const float* Rw    = (const float*)d_in[10];
  const float* Rb    = (const float*)d_in[11];
  const float* Qw    = (const float*)d_in[12];
  const float* Qb    = (const float*)d_in[13];
  const float* Ow    = (const float*)d_in[14];
  const float* Ob    = (const float*)d_in[15];
  const float* fc1w  = (const float*)d_in[16];
  const float* fc1b  = (const float*)d_in[17];
  const float* fc2w  = (const float*)d_in[18];
  const float* fc2b  = (const float*)d_in[19];
  float* out = (float*)d_out;

  // Workspace layout (all offsets 256B-aligned). Peak ~162 MB via aliasing.
  char* ws = (char*)d_ws;
  half_t* c1w16  = (half_t*)(ws + 0);                       //   8,192 B
  half_t* c2w16  = (half_t*)(ws + 8192);                    //  73,728 B
  half_t* Rw16   = (half_t*)(ws + 81920);                   //  93,184 B
  half_t* Ow16   = (half_t*)(ws + 175104);                  // 106,496 B
  half_t* Qw16   = (half_t*)(ws + 281600);                  // 946,176 B
  float*  stats1 = (float*)(ws + 1227776);                  //     512 B
  float*  stats2 = (float*)(ws + 1228288);                  //     512 B
  half_t* preBN1 = (half_t*)(ws + (size_t)(2u << 20));      // 128 MB (1024*64*1024)
  half_t* preBN2 = preBN1;                                  // alias: preBN1 dead after pool1
  half_t* feats  = (half_t*)(ws + (size_t)(2u << 20) + ((size_t)64 << 20)); // 8 MB, past preBN2
  half_t* pooled1= (half_t*)(ws + (size_t)(2u << 20) + ((size_t)128 << 20)); // 32 MB

  // 0) weight conversion + stat zeroing (graph-capturable)
  convert_weights_kernel<<<(S_TOTAL + 255) / 256, 256, 0, stream>>>(
      c1w, c2w, Rw, Ow, Qw, c1w16, c2w16, Rw16, Ow16, Qw16);
  hipMemsetAsync(stats1, 0, 1024, stream);  // stats1 + stats2 contiguous

  // 1) conv1 (implicit GEMM, WMMA) + BN stats
  conv1_kernel<<<1024, 256, 0, stream>>>(x1, x2, c1w16, c1b, preBN1, stats1);

  // 2) BN + ReLU + MaxPool -> (1024,64,16,16)
  {
    const int total = 1024 * 64 * 16 * 16;
    bn_relu_pool_kernel<<<(total + 255) / 256, 256, 0, stream>>>(
        preBN1, pooled1, stats1, bn1g, bn1b, 1.f / (1024.f * 1024.f), 32, 16, total);
  }

  // 3) conv2 (implicit GEMM, WMMA, K=576) + BN stats
  conv2_kernel<<<1024, 256, 0, stream>>>(pooled1, c2w16, c2b, preBN2, stats2);

  // 4) BN + ReLU + MaxPool -> feats (1024, 4096) f16
  {
    const int total = 1024 * 64 * 8 * 8;
    bn_relu_pool_kernel<<<(total + 255) / 256, 256, 0, stream>>>(
        preBN2, feats, stats2, bn2g, bn2b, 1.f / (1024.f * 256.f), 16, 8, total);
  }

  // 5) sequential RNN (T=64) + head, single workgroup, WMMA GEMMs
  rnn_head_kernel<<<1, 224, 0, stream>>>(feats, Rw16, Qw16, Ow16, Rb, Qb, Ob,
                                         fc1w, fc1b, fc2w, fc2b, out);
  (void)in_sizes; (void)n_in; (void)out_size; (void)ws_size;
}
